// AdaptiveNet_SLSTM_74268574482513
// MI455X (gfx1250) — matrix operand order repaired and checked by reading, not compiled, not run on hardware
//
#include <hip/hip_runtime.h>
#include <hip/hip_bf16.h>

typedef __attribute__((ext_vector_type(16))) __bf16 v16bf;
typedef __attribute__((ext_vector_type(8)))  float  v8f;

#define AS1 __attribute__((address_space(1)))

#define HD      128      // hidden size
#define NGATE   512      // 4*H
#define NSTEPS  128      // scan length (B)
#define NROWS   1024     // inner batch (T)
#define NT_G    32       // gate n-tiles (512/16)
#define KC_IN   2        // Wih1: K=42 padded to 64 -> 2 k-chunks of 32
#define KC_H    4        // K=128 -> 4 k-chunks
#define FRAG_EL 512      // bf16 elements per 16x16x32 B fragment (1KB)

// workspace byte offsets
#define WS_W1IH 0
#define WS_W1HH (WS_W1IH + NT_G*KC_IN*FRAG_EL*2)   //  64KB
#define WS_W2IH (WS_W1HH + NT_G*KC_H*FRAG_EL*2)    // +128KB
#define WS_W2HH (WS_W2IH + NT_G*KC_H*FRAG_EL*2)    // +128KB
#define WS_FC   (WS_W2HH + NT_G*KC_H*FRAG_EL*2)    // +128KB
#define WS_B1   (WS_FC   + 1*KC_H*FRAG_EL*2)       // +4KB
#define WS_B2   (WS_B1   + NGATE*4)

// Opaque (per-evaluation) GLOBAL-addrspace pointer: defeats LICM/CSE on the
// per-step weight stream (no register hoisting -> no spills) while still
// lowering to plain saddr-form global_load_b128 (LOADcnt only, decoupled from
// the DScnt domain used by the LDS A-fragment staging).
#define OPAQUE_GPTR(T, p) ({ unsigned long long _v = (unsigned long long)(p); \
                             asm volatile("" : "+s"(_v));                     \
                             (AS1 const T*)_v; })

// ---------------------------------------------------------------------------
// Pack fp32 weight matrix W[N][K] (GEMM is z = A @ W^T, so B[k][n] = W[n][k])
// into bf16 WMMA B-fragment layout: frag index = nt*KC + kc, per-lane 32B.
// Slot e of lane L holds B[kc*32 + e + (e>=8?8:0) + (L>=16?8:0)][nt*16 + (L&15)].
// ---------------------------------------------------------------------------
__global__ void pack_wmma_b(const float* __restrict__ src, __bf16* __restrict__ dst,
                            int N, int K, int KC) {
  const int frag = blockIdx.x;
  const int nt = frag / KC, kc = frag % KC;
  const int lane = threadIdx.x;
  const int n = nt * 16 + (lane & 15);
  const int kbase = kc * 32 + ((lane >= 16) ? 8 : 0);
  __bf16* p = dst + frag * FRAG_EL + lane * 16;
#pragma unroll
  for (int e = 0; e < 16; ++e) {
    int k = kbase + e + ((e >= 8) ? 8 : 0);
    float v = (k < K && n < N) ? src[n * K + k] : 0.0f;
    p[e] = (__bf16)v;
  }
}

__global__ void combine_bias(const float* __restrict__ a, const float* __restrict__ b,
                             float* __restrict__ out) {
  int i = blockIdx.x * blockDim.x + threadIdx.x;
  if (i < NGATE) out[i] = a[i] + b[i];
}

// ---------------------------------------------------------------------------
// Device helpers
// ---------------------------------------------------------------------------
__device__ __forceinline__ float sigm_(float x) { return 1.0f / (1.0f + __expf(-x)); }
__device__ __forceinline__ float tanh_(float x) {
  float e = __expf(-2.0f * fabsf(x));
  float r = (1.0f - e) / (1.0f + e);
  return copysignf(r, x);
}

__device__ __forceinline__ v16bf load_bfrag_g(AS1 const __bf16* base, int frag, int lane) {
  return *(AS1 const v16bf*)(base + frag * FRAG_EL + lane * 16);
}
__device__ __forceinline__ v16bf load_bfrag(const __bf16* base, int frag, int lane) {
  return *(const v16bf*)(base + frag * FRAG_EL + lane * 16);
}

// Build 16-bit A fragment (16x32 tile) from bf16 LDS staging [16 rows][stride].
// ISA A layout: lane L -> M = L&15; slot e -> K = e + (e>=8?8:0) + (L>=16?8:0).
__device__ __forceinline__ v16bf afrag_lds(const __bf16* st, int stride, int lane, int kc) {
  v16bf v;
  const int m = lane & 15;
  const int b8 = (lane >= 16) ? 8 : 0;
#pragma unroll
  for (int j = 0; j < 8; ++j) {
    int k = kc * 32 + 2 * j + ((j >= 4) ? 8 : 0) + b8;
    v[2 * j]     = st[m * stride + k];
    v[2 * j + 1] = st[m * stride + k + 1];
  }
  return v;
}

// ---------------------------------------------------------------------------
// Main recurrent kernel: one wave32 per 16-row tile, full 128-step recurrence.
// ---------------------------------------------------------------------------
__global__ __launch_bounds__(32)
void slstm_main(const float* __restrict__ x,
                const __bf16* __restrict__ w1ih, const __bf16* __restrict__ w1hh,
                const __bf16* __restrict__ w2ih, const __bf16* __restrict__ w2hh,
                const __bf16* __restrict__ wfc,
                const float* __restrict__ bias1, const float* __restrict__ bias2,
                const float* __restrict__ thr1p, const float* __restrict__ thr2p,
                const float* __restrict__ fcb, float* __restrict__ out) {
  const int tile = blockIdx.x;        // 0..63, rows tile*16 .. tile*16+15
  const int lane = threadIdx.x;       // 0..31
  const float thr1 = thr1p[0], thr2 = thr2p[0];

  __shared__ __bf16 s_in[16 * 64];    // delta-mod spikes, K padded 42->64
  __shared__ __bf16 s_m1[16 * HD];    // mem1 (bf16 A staging)
  __shared__ __bf16 s_s1[16 * HD];    // spk1
  __shared__ __bf16 s_m2[16 * HD];    // mem2
  __shared__ float  s_c1[16 * HD];    // syn1 (fp32)
  __shared__ float  s_c2[16 * HD];    // syn2
  __shared__ float  s_acc[16 * HD];   // sum of mem2 over steps

  for (int i = lane; i < 16 * 64; i += 32) s_in[i] = (__bf16)0.0f;
  for (int i = lane; i < 16 * HD; i += 32) {
    s_m1[i] = (__bf16)0.0f; s_m2[i] = (__bf16)0.0f;
    s_c1[i] = 0.0f; s_c2[i] = 0.0f; s_acc[i] = 0.0f;
  }

  const float THR0 = 9.9893e-06f, THR1v = 2.9968e-05f, THR2v = 5.9936e-05f;
  const int nn = lane & 15;
  const int mhalf = (lane >= 16) ? 8 : 0;

#pragma unroll 1
  for (int b = 0; b < NSTEPS; ++b) {
    // Per-iteration opaque global pointers (weights streamed from L2 each step).
    AS1 const __bf16* w1ih_s = OPAQUE_GPTR(__bf16, w1ih);
    AS1 const __bf16* w1hh_s = OPAQUE_GPTR(__bf16, w1hh);
    AS1 const __bf16* w2ih_s = OPAQUE_GPTR(__bf16, w2ih);
    AS1 const __bf16* w2hh_s = OPAQUE_GPTR(__bf16, w2hh);
    AS1 const float*  b1_s   = OPAQUE_GPTR(float, bias1);
    AS1 const float*  b2_s   = OPAQUE_GPTR(float, bias2);

    // ---- delta-modulation spikes for this step into s_in (16 rows x 42) ----
#pragma unroll
    for (int it = 0; it < 7; ++it) {
      int idx = it * 32 + lane;                       // (t_local, feature) pairs
      if (idx < 16 * 14) {
        int tl = idx / 14, f = idx % 14;
        int tg = tile * 16 + tl;
        float d = 0.0f;
        if (tg > 0)
          d = x[(b * NROWS + tg) * 14 + f] - x[(b * NROWS + tg - 1) * 14 + f];
        s_in[tl * 64 + f * 3 + 0] = (__bf16)((d > THR0)  ? 1.0f : 0.0f);
        s_in[tl * 64 + f * 3 + 1] = (__bf16)((d > THR1v) ? 1.0f : 0.0f);
        s_in[tl * 64 + f * 3 + 2] = (__bf16)((d > THR2v) ? 1.0f : 0.0f);
      }
    }

    // ---- layer 1 A fragments (input spikes + old mem1) ----
    v16bf aIn[KC_IN], aM1[KC_H];
#pragma unroll
    for (int kc = 0; kc < KC_IN; ++kc) aIn[kc] = afrag_lds(s_in, 64, lane, kc);
#pragma unroll
    for (int kc = 0; kc < KC_H; ++kc) aM1[kc] = afrag_lds(s_m1, HD, lane, kc);

    // ---- layer 1: per 16-wide h-tile, 4 gate tiles via WMMA, then fused cell ----
#pragma unroll
    for (int j = 0; j < 8; ++j) {
      v8f cg[4];
#pragma unroll
      for (int g = 0; g < 4; ++g) {
        int nt = g * 8 + j;
        float bv = b1_s[nt * 16 + nn];
        v8f c;
#pragma unroll
        for (int r = 0; r < 8; ++r) c[r] = bv;
#pragma unroll
        for (int kc = 0; kc < KC_IN; ++kc) {
          v16bf bb = load_bfrag_g(w1ih_s, nt * KC_IN + kc, lane);
          c = __builtin_amdgcn_wmma_f32_16x16x32_bf16(false, aIn[kc], false, bb,
                                                      (short)0, c, false, false);
        }
#pragma unroll
        for (int kc = 0; kc < KC_H; ++kc) {
          v16bf bb = load_bfrag_g(w1hh_s, nt * KC_H + kc, lane);
          c = __builtin_amdgcn_wmma_f32_16x16x32_bf16(false, aM1[kc], false, bb,
                                                      (short)0, c, false, false);
        }
        cg[g] = c;
      }
#pragma unroll
      for (int r = 0; r < 8; ++r) {
        int m = r + mhalf, h = j * 16 + nn;
        float zi = cg[0][r], zf = cg[1][r], zg = cg[2][r], zo = cg[3][r];
        float cold = s_c1[m * HD + h];
        float mold = (float)s_m1[m * HD + h];
        float reset = ((mold - thr1) > 0.0f) ? 1.0f : 0.0f;
        float c2 = sigm_(zf) * cold + sigm_(zi) * tanh_(zg);
        float h2 = sigm_(zo) * tanh_(c2);
        float mnew = h2 - reset * thr1;
        s_c1[m * HD + h] = c2;
        s_m1[m * HD + h] = (__bf16)mnew;
        s_s1[m * HD + h] = (__bf16)(((mnew - thr1) > 0.0f) ? 1.0f : 0.0f);
      }
    }

    // ---- layer 2 A fragments (spk1 + old mem2) ----
    v16bf aS1[KC_H], aM2[KC_H];
#pragma unroll
    for (int kc = 0; kc < KC_H; ++kc) {
      aS1[kc] = afrag_lds(s_s1, HD, lane, kc);
      aM2[kc] = afrag_lds(s_m2, HD, lane, kc);
    }

#pragma unroll
    for (int j = 0; j < 8; ++j) {
      v8f cg[4];
#pragma unroll
      for (int g = 0; g < 4; ++g) {
        int nt = g * 8 + j;
        float bv = b2_s[nt * 16 + nn];
        v8f c;
#pragma unroll
        for (int r = 0; r < 8; ++r) c[r] = bv;
#pragma unroll
        for (int kc = 0; kc < KC_H; ++kc) {
          v16bf bb = load_bfrag_g(w2ih_s, nt * KC_H + kc, lane);
          c = __builtin_amdgcn_wmma_f32_16x16x32_bf16(false, aS1[kc], false, bb,
                                                      (short)0, c, false, false);
        }
#pragma unroll
        for (int kc = 0; kc < KC_H; ++kc) {
          v16bf bb = load_bfrag_g(w2hh_s, nt * KC_H + kc, lane);
          c = __builtin_amdgcn_wmma_f32_16x16x32_bf16(false, aM2[kc], false, bb,
                                                      (short)0, c, false, false);
        }
        cg[g] = c;
      }
#pragma unroll
      for (int r = 0; r < 8; ++r) {
        int m = r + mhalf, h = j * 16 + nn;
        float zi = cg[0][r], zf = cg[1][r], zg = cg[2][r], zo = cg[3][r];
        float cold = s_c2[m * HD + h];
        float mold = (float)s_m2[m * HD + h];
        float reset = ((mold - thr2) > 0.0f) ? 1.0f : 0.0f;
        float c2 = sigm_(zf) * cold + sigm_(zi) * tanh_(zg);
        float h2 = sigm_(zo) * tanh_(c2);
        float mnew = h2 - reset * thr2;
        s_c2[m * HD + h] = c2;
        s_m2[m * HD + h] = (__bf16)mnew;
        s_acc[m * HD + h] += mnew;
      }
    }
  } // steps

  // ---- final: mean over steps, then FC (N=8 padded to 16) via 4 WMMAs ----
#pragma unroll
  for (int j = 0; j < 8; ++j)
#pragma unroll
    for (int r = 0; r < 8; ++r) {
      int m = r + mhalf, h = j * 16 + nn;
      s_m1[m * HD + h] = (__bf16)(s_acc[m * HD + h] * (1.0f / (float)NSTEPS));
    }

  float bv = fcb[(nn < 8) ? nn : 0];
  v8f co;
#pragma unroll
  for (int r = 0; r < 8; ++r) co[r] = (nn < 8) ? bv : 0.0f;
#pragma unroll
  for (int kc = 0; kc < KC_H; ++kc) {
    v16bf aA = afrag_lds(s_m1, HD, lane, kc);
    v16bf bb = load_bfrag(wfc, kc, lane);
    co = __builtin_amdgcn_wmma_f32_16x16x32_bf16(false, aA, false, bb,
                                                 (short)0, co, false, false);
  }
#pragma unroll
  for (int r = 0; r < 8; ++r) {
    int m = r + mhalf;
    if (nn < 8) out[(tile * 16 + m) * 8 + nn] = co[r];
  }
}

// ---------------------------------------------------------------------------
extern "C" void kernel_launch(void* const* d_in, const int* in_sizes, int n_in,
                              void* d_out, int out_size, void* d_ws, size_t ws_size,
                              hipStream_t stream) {
  const float* x    = (const float*)d_in[0];
  const float* Wih1 = (const float*)d_in[1];
  const float* Whh1 = (const float*)d_in[2];
  const float* bih1 = (const float*)d_in[3];
  const float* bhh1 = (const float*)d_in[4];
  const float* thr1 = (const float*)d_in[5];
  const float* Wih2 = (const float*)d_in[6];
  const float* Whh2 = (const float*)d_in[7];
  const float* bih2 = (const float*)d_in[8];
  const float* bhh2 = (const float*)d_in[9];
  const float* thr2 = (const float*)d_in[10];
  const float* fcW  = (const float*)d_in[11];
  const float* fcb  = (const float*)d_in[12];

  char* ws = (char*)d_ws;
  __bf16* w1ih = (__bf16*)(ws + WS_W1IH);
  __bf16* w1hh = (__bf16*)(ws + WS_W1HH);
  __bf16* w2ih = (__bf16*)(ws + WS_W2IH);
  __bf16* w2hh = (__bf16*)(ws + WS_W2HH);
  __bf16* wfc  = (__bf16*)(ws + WS_FC);
  float*  b1   = (float*)(ws + WS_B1);
  float*  b2   = (float*)(ws + WS_B2);

  pack_wmma_b<<<NT_G * KC_IN, 32, 0, stream>>>(Wih1, w1ih, NGATE, 42, KC_IN);
  pack_wmma_b<<<NT_G * KC_H, 32, 0, stream>>>(Whh1, w1hh, NGATE, HD, KC_H);
  pack_wmma_b<<<NT_G * KC_H, 32, 0, stream>>>(Wih2, w2ih, NGATE, HD, KC_H);
  pack_wmma_b<<<NT_G * KC_H, 32, 0, stream>>>(Whh2, w2hh, NGATE, HD, KC_H);
  pack_wmma_b<<<1 * KC_H, 32, 0, stream>>>(fcW, wfc, 8, HD, KC_H);
  combine_bias<<<2, 256, 0, stream>>>(bih1, bhh1, b1);
  combine_bias<<<2, 256, 0, stream>>>(bih2, bhh2, b2);

  slstm_main<<<NROWS / 16, 32, 0, stream>>>(x, w1ih, w1hh, w2ih, w2hh, wfc,
                                            b1, b2, thr1, thr2, fcb,
                                            (float*)d_out);
}
